// MultiHeadAttention_16578573762927
// MI455X (gfx1250) — matrix operand order repaired and checked by reading, not compiled, not run on hardware
//
#include <hip/hip_runtime.h>
#include <hip/hip_bf16.h>

#define DEV __device__ __forceinline__

constexpr int Bc  = 2;
constexpr int Sc  = 2048;
constexpr int Dc  = 2048;
constexpr int Hc  = 16;
constexpr int HDc = 128;
constexpr int BSc = Bc * Sc;   // 4096 flattened rows

typedef __attribute__((ext_vector_type(16))) __bf16 v16bf;
typedef __attribute__((ext_vector_type(8)))  __bf16 v8bf;
typedef __attribute__((ext_vector_type(8)))  float  v8f;

DEV v16bf cat8(v8bf lo, v8bf hi) {
  return __builtin_shufflevector(lo, hi, 0,1,2,3,4,5,6,7,8,9,10,11,12,13,14,15);
}
DEV v8f wmma_bf16f32(v16bf a, v16bf b, v8f c) {
  // D(16x16,f32) = A(16x32,bf16) * B(32x16,bf16) + C
  return __builtin_amdgcn_wmma_f32_16x16x32_bf16(false, a, false, b, (short)0, c, false, false);
}

// CDNA5 async global->LDS copy (ASYNCcnt-tracked, bypasses VGPRs).
// LDS address comes from the vdst VGPR; low 32 bits of a generic LDS pointer are the
// LDS byte offset (ISA 10.2).
DEV void async_ld128(const void* lds_dst, const void* gsrc) {
  unsigned int  lo = (unsigned int)(unsigned long long)lds_dst;
  unsigned long long ga = (unsigned long long)gsrc;
  asm volatile("global_load_async_to_lds_b128 %0, %1, off"
               :: "v"(lo), "v"(ga) : "memory");
}
DEV void wait_async0() {
  asm volatile("s_wait_asynccnt 0" ::: "memory");
}

// ---------------------------------------------------------------------------
// f32 -> bf16 cast
// ---------------------------------------------------------------------------
__global__ void cvt_f32_bf16(const float* __restrict__ s, __bf16* __restrict__ d, int n) {
  int i = (blockIdx.x * blockDim.x + threadIdx.x) * 4;
  if (i + 3 < n) {
    float4 v = *(const float4*)(s + i);
    d[i+0] = (__bf16)v.x; d[i+1] = (__bf16)v.y;
    d[i+2] = (__bf16)v.z; d[i+3] = (__bf16)v.w;
  } else {
    for (; i < n; ++i) d[i] = (__bf16)s[i];
  }
}

// ---------------------------------------------------------------------------
// RoPE tables: ct/st[s*64 + i] = cos/sin(s * 10000^(-i/64)),  s<2048, i<64
// ---------------------------------------------------------------------------
__global__ void rope_tables(float* __restrict__ ct, float* __restrict__ st) {
  int idx = blockIdx.x * blockDim.x + threadIdx.x;
  if (idx >= Sc * 64) return;
  int s = idx >> 6, i = idx & 63;
  float ang = (float)s * __expf(-(float)i * 0.14391156831212791f); // ln(1e4)/64
  float sn, cs;
  sincosf(ang, &sn, &cs);
  ct[idx] = cs;
  st[idx] = sn;
}

// ---------------------------------------------------------------------------
// GEMM: out[M,N] = A[M,K] @ W[N,K]^T   (NT, both operands K-contiguous)
// mode 0: Q proj  -> RoPE + 1/sqrt(hd), bf16 out [B,H,S,HD]
// mode 1: K proj  -> RoPE,              bf16 out [B,H,S,HD]
// mode 2: V proj  ->                    bf16 out [B,H,S,HD]
// mode 3: O proj  ->                    f32  out [B,S,D] (d_out)
// 128x128 tile / WG, 8 waves x (16x128). K staged 64-wide, double-buffered
// LDS fed by global_load_async_to_lds_b128.
// ---------------------------------------------------------------------------
__global__ __launch_bounds__(256) void gemm_proj(
    const __bf16* __restrict__ A, const __bf16* __restrict__ Wbase, size_t wstride,
    __bf16* __restrict__ outb_base, size_t obstride,
    float* __restrict__ outf,
    const float* __restrict__ Ct, const float* __restrict__ St, int mode_base)
{
  __shared__ __bf16 sA[2][128][72];   // 144B rows (16B multiple), banks spread
  __shared__ __bf16 sB[2][128][72];

  const int mode = mode_base + (int)blockIdx.z;
  const __bf16* W = Wbase + wstride * blockIdx.z;
  __bf16* outb = outb_base ? outb_base + obstride * blockIdx.z : nullptr;

  const int gm0 = blockIdx.x * 128;
  const int n0  = blockIdx.y * 128;
  const int wave = threadIdx.x >> 5;
  const int lane = threadIdx.x & 31;
  const int hl   = lane >> 4;
  const int l16  = lane & 15;

  const int ldrow = threadIdx.x >> 1;       // 0..127
  const int ldc0  = (threadIdx.x & 1) * 32; // 0 or 32

  auto stage = [&](int buf, int k0) {
    const __bf16* gA = A + (size_t)(gm0 + ldrow) * Dc + k0 + ldc0;
    const __bf16* gW = W + (size_t)(n0 + ldrow) * Dc + k0 + ldc0;
    #pragma unroll
    for (int c = 0; c < 4; ++c) {
      async_ld128(&sA[buf][ldrow][ldc0 + c*8], gA + c*8);
      async_ld128(&sB[buf][ldrow][ldc0 + c*8], gW + c*8);
    }
  };

  v8f zero = {};
  v8f acc[8];
  #pragma unroll
  for (int j = 0; j < 8; ++j) acc[j] = zero;

  stage(0, 0);
  wait_async0();
  __syncthreads();

  for (int k0 = 0; k0 < Dc; k0 += 64) {
    const int cur = (k0 >> 6) & 1;
    if (k0 + 64 < Dc) stage(cur ^ 1, k0 + 64);   // async prefetch next stage

    #pragma unroll
    for (int t = 0; t < 2; ++t) {
      // A fragment (16x32): lane<16 -> K {0..7,16..23}; lane>=16 -> {8..15,24..31}
      v8bf a0 = *(const v8bf*)&sA[cur][wave*16 + l16][t*32 + hl*8];
      v8bf a1 = *(const v8bf*)&sA[cur][wave*16 + l16][t*32 + hl*8 + 16];
      v16bf af = cat8(a0, a1);
      v16bf bfr[8];
      #pragma unroll
      for (int j = 0; j < 8; ++j) {
        v8bf b0 = *(const v8bf*)&sB[cur][j*16 + l16][t*32 + hl*8];
        v8bf b1 = *(const v8bf*)&sB[cur][j*16 + l16][t*32 + hl*8 + 16];
        bfr[j] = cat8(b0, b1);
      }
      #pragma unroll
      for (int j = 0; j < 8; ++j) acc[j] = wmma_bf16f32(af, bfr[j], acc[j]);
    }

    wait_async0();     // next-stage LDS writes landed
    __syncthreads();   // everyone done reading cur / writing nxt
  }

  // Epilogue. C layout: element r of acc[j] is M = r + 8*hl, N = j*16 + l16.
  const int rowbase = gm0 + wave*16 + hl*8;
  #pragma unroll
  for (int r = 0; r < 8; ++r) {
    int gm   = rowbase + r;
    int spos = gm & (Sc - 1);
    int bidx = gm >> 11;          // S = 2048
    #pragma unroll
    for (int j = 0; j < 8; ++j) {
      float v = acc[j][r];
      float outv;
      if (mode <= 1) {
        // RoPE: partner q[d^64] lives in acc[j^4], same lane, same element r
        float p = acc[j ^ 4][r];
        int d = j*16 + l16;
        int ti = (spos << 6) + (d & 63);
        float cs = Ct[ti];
        float sn = St[ti];
        float rot = (d < 64) ? -p : p;
        outv = v * cs + rot * sn;
        if (mode == 0) outv *= 0.08838834764831845f;  // hd^-0.5 folded into Q
      } else {
        outv = v;
      }
      if (mode < 3) {
        size_t idx = ((((size_t)bidx * Hc) + (n0 >> 7)) * Sc + spos) * (size_t)HDc + (j*16 + l16);
        outb[idx] = (__bf16)outv;
      } else {
        outf[(size_t)gm * Dc + n0 + j*16 + l16] = outv;
      }
    }
  }
}

// ---------------------------------------------------------------------------
// Flash attention over bf16 Q/K/V [B,H,S,HD]; causal; Q pre-scaled + roped.
// 128 query rows per workgroup (8 waves x 16 rows); key blocks of 128.
// K tile staged via async global->LDS; V transposed through VGPRs.
// ---------------------------------------------------------------------------
__global__ __launch_bounds__(256) void flash_attn(
    const __bf16* __restrict__ Qb, const __bf16* __restrict__ Kb,
    const __bf16* __restrict__ Vb, __bf16* __restrict__ Ab)
{
  __shared__ __bf16 sK [128][136];     // [key][d]
  __shared__ __bf16 sVT[128][136];     // [d][key]
  __shared__ __bf16 sP [8][16][136];   // per-wave P scratch

  const int qt = blockIdx.x;
  const int h  = blockIdx.y;
  const int b  = blockIdx.z;
  const int q0 = qt * 128;
  const int wave = threadIdx.x >> 5;
  const int lane = threadIdx.x & 31;
  const int hl   = lane >> 4;
  const int l16  = lane & 15;

  const size_t headbase = (((size_t)b * Hc) + h) * (size_t)Sc * HDc;

  // Q fragments stay in registers: wave's 16 rows, HD=128 = 4 k-steps of 32
  const __bf16* qrp = Qb + headbase + (size_t)(q0 + wave*16 + l16) * HDc;
  v16bf qf[4];
  #pragma unroll
  for (int t = 0; t < 4; ++t) {
    v8bf lo = *(const v8bf*)(qrp + t*32 + hl*8);
    v8bf hi = *(const v8bf*)(qrp + t*32 + hl*8 + 16);
    qf[t] = cat8(lo, hi);
  }

  v8f zero = {};
  float m[8], l[8];
  v8f o[8];
  #pragma unroll
  for (int r = 0; r < 8; ++r) { m[r] = -1e30f; l[r] = 0.0f; }
  #pragma unroll
  for (int j = 0; j < 8; ++j) o[j] = zero;

  const int ldrow = threadIdx.x >> 1;        // 0..127
  const int ldcb  = (threadIdx.x & 1) * 64;  // 0 or 64

  for (int kb = 0; kb <= qt; ++kb) {         // causal: keys up to q-tile
    const int k0 = kb * 128;
    __syncthreads();
    {
      // K tile: async DMA straight into LDS
      const __bf16* gk = Kb + headbase + (size_t)(k0 + ldrow) * HDc + ldcb;
      #pragma unroll
      for (int c = 0; c < 8; ++c)
        async_ld128(&sK[ldrow][ldcb + c*8], gk + c*8);
      // V tile: through VGPRs, transposed into sVT
      const __bf16* gv = Vb + headbase + (size_t)(k0 + ldrow) * HDc + ldcb;
      #pragma unroll
      for (int c = 0; c < 8; ++c) {
        v8bf vv = *(const v8bf*)(gv + c*8);
        #pragma unroll
        for (int i = 0; i < 8; ++i) sVT[ldcb + c*8 + i][ldrow] = vv[i];
      }
    }
    wait_async0();
    __syncthreads();

    // S = Q K^T : per wave a 16x128 score strip = 8 C-tiles, 4 k-steps each
    v8f sacc[8];
    #pragma unroll
    for (int kn = 0; kn < 8; ++kn) sacc[kn] = zero;
    #pragma unroll
    for (int kn = 0; kn < 8; ++kn) {
      #pragma unroll
      for (int t = 0; t < 4; ++t) {
        v8bf b0 = *(const v8bf*)&sK[kn*16 + l16][t*32 + hl*8];
        v8bf b1 = *(const v8bf*)&sK[kn*16 + l16][t*32 + hl*8 + 16];
        sacc[kn] = wmma_bf16f32(qf[t], cat8(b0, b1), sacc[kn]);
      }
    }

    if (kb == qt) {  // diagonal block: causal mask
      #pragma unroll
      for (int kn = 0; kn < 8; ++kn) {
        int kcol = k0 + kn*16 + l16;
        #pragma unroll
        for (int r = 0; r < 8; ++r) {
          int qr = q0 + wave*16 + hl*8 + r;
          if (kcol > qr) sacc[kn][r] = -1e30f;
        }
      }
    }

    // Online softmax: row = (r, lane-half); 16-lane xor reductions per half
    #pragma unroll
    for (int r = 0; r < 8; ++r) {
      float mx = sacc[0][r];
      #pragma unroll
      for (int kn = 1; kn < 8; ++kn) mx = fmaxf(mx, sacc[kn][r]);
      mx = fmaxf(mx, __shfl_xor(mx, 1, 32));
      mx = fmaxf(mx, __shfl_xor(mx, 2, 32));
      mx = fmaxf(mx, __shfl_xor(mx, 4, 32));
      mx = fmaxf(mx, __shfl_xor(mx, 8, 32));
      float mnew  = fmaxf(m[r], mx);
      float alpha = __expf(m[r] - mnew);
      float rs = 0.0f;
      #pragma unroll
      for (int kn = 0; kn < 8; ++kn) {
        float p = __expf(sacc[kn][r] - mnew);
        sacc[kn][r] = p;
        rs += p;
      }
      rs += __shfl_xor(rs, 1, 32);
      rs += __shfl_xor(rs, 2, 32);
      rs += __shfl_xor(rs, 4, 32);
      rs += __shfl_xor(rs, 8, 32);
      l[r] = l[r] * alpha + rs;
      m[r] = mnew;
      #pragma unroll
      for (int j = 0; j < 8; ++j) o[j][r] *= alpha;
    }

    // P: C-layout regs -> LDS -> A-layout fragments
    #pragma unroll
    for (int kn = 0; kn < 8; ++kn)
      #pragma unroll
      for (int r = 0; r < 8; ++r)
        sP[wave][hl*8 + r][kn*16 + l16] = (__bf16)sacc[kn][r];
    __syncthreads();

    // O += P * V : A = P (16 x 128 keys), B = V (keys x d) from transposed tile
    #pragma unroll
    for (int t = 0; t < 4; ++t) {
      v8bf p0 = *(const v8bf*)&sP[wave][l16][t*32 + hl*8];
      v8bf p1 = *(const v8bf*)&sP[wave][l16][t*32 + hl*8 + 16];
      v16bf pf = cat8(p0, p1);
      #pragma unroll
      for (int j = 0; j < 8; ++j) {
        v8bf b0 = *(const v8bf*)&sVT[j*16 + l16][t*32 + hl*8];
        v8bf b1 = *(const v8bf*)&sVT[j*16 + l16][t*32 + hl*8 + 16];
        o[j] = wmma_bf16f32(pf, cat8(b0, b1), o[j]);
      }
    }
  }

  // Normalize and store attn output as bf16 [B,S,D] (heads re-interleaved)
  #pragma unroll
  for (int r = 0; r < 8; ++r) {
    int srow = q0 + wave*16 + hl*8 + r;
    float inv = 1.0f / l[r];
    #pragma unroll
    for (int j = 0; j < 8; ++j) {
      size_t idx = ((size_t)b * Sc + srow) * (size_t)Dc + h * HDc + j*16 + l16;
      Ab[idx] = (__bf16)(o[j][r] * inv);
    }
  }
}

// ---------------------------------------------------------------------------
extern "C" void kernel_launch(void* const* d_in, const int* in_sizes, int n_in,
                              void* d_out, int out_size, void* d_ws, size_t ws_size,
                              hipStream_t stream) {
  (void)in_sizes; (void)n_in; (void)out_size;
  const float* x  = (const float*)d_in[0];
  const float* Wq = (const float*)d_in[1];
  const float* Wk = (const float*)d_in[2];
  const float* Wv = (const float*)d_in[3];
  const float* Wo = (const float*)d_in[4];
  float* out = (float*)d_out;

  const size_t NX = (size_t)BSc * Dc;            // 8M elems: x (bf16)
  const size_t NW = (size_t)Dc * Dc;             // 4M elems: each weight
  const size_t NQ = (size_t)Bc * Hc * Sc * HDc;  // 8M elems: Q/K/V/attn-out
  const size_t bf16_bytes = (NX + 4*NW + 4*NQ) * sizeof(__bf16);  // 112 MiB
  const size_t rope_elems = (size_t)Sc * 64;
  const size_t need = bf16_bytes + 2 * rope_elems * sizeof(float);
  if (ws_size < need) return;

  __bf16* xbf  = (__bf16*)d_ws;
  __bf16* wq   = xbf + NX;
  __bf16* wk   = wq + NW;
  __bf16* wv   = wk + NW;
  __bf16* wo   = wv + NW;
  __bf16* qb   = wo + NW;
  __bf16* kbuf = qb + NQ;
  __bf16* vbuf = kbuf + NQ;
  __bf16* ab   = vbuf + NQ;
  float*  ctab = (float*)((char*)d_ws + bf16_bytes);
  float*  stab = ctab + rope_elems;

  rope_tables<<<(int)((rope_elems + 255) / 256), 256, 0, stream>>>(ctab, stab);

  {
    int n = (int)NX;
    cvt_f32_bf16<<<(n/4 + 255)/256, 256, 0, stream>>>(x, xbf, n);
  }
  {
    int n = (int)NW;
    int g = (n/4 + 255)/256;
    cvt_f32_bf16<<<g, 256, 0, stream>>>(Wq, wq, n);
    cvt_f32_bf16<<<g, 256, 0, stream>>>(Wk, wk, n);
    cvt_f32_bf16<<<g, 256, 0, stream>>>(Wv, wv, n);
    cvt_f32_bf16<<<g, 256, 0, stream>>>(Wo, wo, n);
  }

  // Q/K/V projections + RoPE epilogue (z selects weight/output/mode)
  gemm_proj<<<dim3(BSc/128, Dc/128, 3), 256, 0, stream>>>(
      xbf, wq, NW, qb, NQ, nullptr, ctab, stab, 0);

  // Causal flash attention
  flash_attn<<<dim3(Sc/128, Hc, Bc), 256, 0, stream>>>(qb, kbuf, vbuf, ab);

  // Output projection, f32 result straight to d_out
  gemm_proj<<<dim3(BSc/128, Dc/128, 1), 256, 0, stream>>>(
      ab, wo, 0, nullptr, 0, out, ctab, stab, 3);
}
